// PointNetSetAbstraction_50543175139454
// MI455X (gfx1250) — matrix operand order repaired and checked by reading, not compiled, not run on hardware
//
#include <hip/hip_runtime.h>
#include <hip/hip_bf16.h>
#include <hip/hip_fp16.h>

#define NB     16
#define NPTS   4096
#define NPOINT 1024
#define NSAMPLE 32
#define DFEAT  64
#define CIN1   67      // 3 + D
#define KPAD1  96      // CIN1 padded to multiple of 32
#define C1     64
#define C2     64
#define C3     128
#define RAD2   (0.2f * 0.2f)
#define EPSV   1e-5f

typedef __attribute__((ext_vector_type(16))) _Float16 v16h;
typedef __attribute__((ext_vector_type(8)))  _Float16 v8h;
typedef __attribute__((ext_vector_type(8)))  float    v8f;

static constexpr size_t NTOT = (size_t)NB * NPOINT * NSAMPLE;   // 524288 positions

// Feature / activation layout ("16-channel grouped", WMMA-native):
//   element (channel k, position n) lives at  ((k/16)*NTOT + n)*16 + (k%16)
// so one lane's 16-high B fragment (K = ks+half*16 .. +15, column n) is one
// contiguous 32B v16h load, and one lane's 8 D outputs are one 16B v8h store.

// ---------------------------------------------------------------------------
// Kernel 1: farthest point sampling. One workgroup per batch. xyz in LDS,
// per-thread running min-distance in VGPRs, LDS argmax tree per step.
// ---------------------------------------------------------------------------
__global__ __launch_bounds__(256) void fps_kernel(const float* __restrict__ xyz,
                                                  float* __restrict__ new_xyz,   // ws: (B, NPOINT, 3)
                                                  float* __restrict__ out_xyz)   // out: (B, 3, NPOINT)
{
    __shared__ float sx[NPTS], sy[NPTS], sz[NPTS];
    __shared__ float rval[256];
    __shared__ int   ridx[256];
    __shared__ int   s_far;
    __shared__ float s_c[3];

    const int b = blockIdx.x;
    const int tid = threadIdx.x;
    const float* xb = xyz + (size_t)b * 3 * NPTS;

    for (int i = tid; i < NPTS; i += 256) {
        sx[i] = xb[i];
        sy[i] = xb[NPTS + i];
        sz[i] = xb[2 * NPTS + i];
    }
    float dist[NPTS / 256];
#pragma unroll
    for (int i = 0; i < NPTS / 256; ++i) dist[i] = 1e10f;
    if (tid == 0) s_far = 0;
    __syncthreads();

    for (int s = 0; s < NPOINT; ++s) {
        if (tid == 0) {
            const int far = s_far;
            const float cx = sx[far], cy = sy[far], cz = sz[far];
            s_c[0] = cx; s_c[1] = cy; s_c[2] = cz;
            new_xyz[((size_t)b * NPOINT + s) * 3 + 0] = cx;
            new_xyz[((size_t)b * NPOINT + s) * 3 + 1] = cy;
            new_xyz[((size_t)b * NPOINT + s) * 3 + 2] = cz;
            out_xyz[(size_t)b * 3 * NPOINT + 0 * NPOINT + s] = cx;
            out_xyz[(size_t)b * 3 * NPOINT + 1 * NPOINT + s] = cy;
            out_xyz[(size_t)b * 3 * NPOINT + 2 * NPOINT + s] = cz;
        }
        __syncthreads();
        const float cx = s_c[0], cy = s_c[1], cz = s_c[2];

        float best = -1.0f; int bidx = 0;
#pragma unroll
        for (int i = 0; i < NPTS / 256; ++i) {
            const int p = tid + i * 256;
            const float dx = sx[p] - cx, dy = sy[p] - cy, dz = sz[p] - cz;
            const float d = dx * dx + dy * dy + dz * dz;
            const float nd = fminf(dist[i], d);
            dist[i] = nd;
            if (nd > best) { best = nd; bidx = p; }
        }
        rval[tid] = best; ridx[tid] = bidx;
        __syncthreads();
        for (int off = 128; off > 0; off >>= 1) {
            if (tid < off) {
                const float ov = rval[tid + off];
                const int   oi = ridx[tid + off];
                if (ov > rval[tid] || (ov == rval[tid] && oi < ridx[tid])) {
                    rval[tid] = ov; ridx[tid] = oi;
                }
            }
            __syncthreads();
        }
        if (tid == 0) s_far = ridx[0];
        __syncthreads();
    }
}

// ---------------------------------------------------------------------------
// Kernel 2: ball query + grouping + concat, one wave32 per query point.
// Ballot + prefix-popcount compacts the first NSAMPLE in-radius indices.
// Writes the grouped/concatenated features in 16-channel-grouped f16 layout
// (rows 0..2 = grouped_xyz - center, 3..66 = grouped pts, 67..95 = zero pad),
// one 32B v16h store per channel group.
// ---------------------------------------------------------------------------
__global__ __launch_bounds__(32) void group_kernel(const float* __restrict__ xyz,
                                                   const float* __restrict__ pts,
                                                   const float* __restrict__ new_xyz,
                                                   _Float16* __restrict__ F)
{
    const int g = blockIdx.x;            // b*NPOINT + s
    const int b = g / NPOINT;
    const int lane = threadIdx.x;
    __shared__ int sidx[NSAMPLE];
    __shared__ int s_found;

    const float* xb = xyz + (size_t)b * 3 * NPTS;
    const float cx = new_xyz[(size_t)g * 3 + 0];
    const float cy = new_xyz[(size_t)g * 3 + 1];
    const float cz = new_xyz[(size_t)g * 3 + 2];

    if (lane == 0) s_found = 0;
    __syncthreads();

    for (int base = 0; base < NPTS; base += 32) {
        __syncthreads();
        const int found = s_found;
        if (found >= NSAMPLE) break;
        const int p = base + lane;
        const float dx = xb[p] - cx, dy = xb[NPTS + p] - cy, dz = xb[2 * NPTS + p] - cz;
        const bool in = (dx * dx + dy * dy + dz * dz) <= RAD2;
        const unsigned mask = __builtin_amdgcn_ballot_w32(in);
        const int rank = __popc(mask & ((1u << lane) - 1u));
        if (in && (found + rank) < NSAMPLE) sidx[found + rank] = p;
        if (lane == 0) {
            int nf = found + __popc(mask);
            s_found = nf > NSAMPLE ? NSAMPLE : nf;
        }
    }
    __syncthreads();
    const int found = s_found;
    const int first = (found > 0) ? sidx[0] : 0;
    const int j = (lane < found) ? sidx[lane] : first;

    const size_t n = (size_t)g * NSAMPLE + lane;   // this lane's column
    const float* pb = pts + (size_t)b * DFEAT * NPTS;
    const float d0 = xb[j] - cx, d1 = xb[NPTS + j] - cy, d2 = xb[2 * NPTS + j] - cz;

#pragma unroll
    for (int kg = 0; kg < KPAD1 / 16; ++kg) {
        v16h chunk;
#pragma unroll
        for (int t = 0; t < 16; ++t) {
            const int c = kg * 16 + t;
            float v;
            if (c == 0)        v = d0;
            else if (c == 1)   v = d1;
            else if (c == 2)   v = d2;
            else if (c < CIN1) v = pb[(size_t)(c - 3) * NPTS + j];
            else               v = 0.0f;
            chunk[t] = (_Float16)v;
        }
        *(v16h*)(F + ((size_t)kg * NTOT + n) * 16) = chunk;
    }
}

// ---------------------------------------------------------------------------
// Kernel 3: weight prep — f32 (Cout,Cin) -> zero-padded f16 (Cout,KP).
// ---------------------------------------------------------------------------
__global__ __launch_bounds__(256) void wprep_kernel(const float* __restrict__ W,
                                                    _Float16* __restrict__ W16,
                                                    int Cout, int Cin, int KP)
{
    const int i = blockIdx.x * 256 + threadIdx.x;
    if (i >= Cout * KP) return;
    const int row = i / KP, k = i % KP;
    W16[i] = (k < Cin) ? (_Float16)W[(size_t)row * Cin + k] : (_Float16)0.0f;
}

// ---------------------------------------------------------------------------
// Kernel 4: conv as WMMA GEMM. One wave owns one 16-position column strip and
// computes ALL Cout tiles for it (TM tiles), so each B fragment is loaded once
// per k-step and reused across TM WMMAs. Fully unrolled: TM*KP/32 static
// v_wmma_f32_16x16x32_f16. A fragments: 2x b128 from padded f16 weights.
// B fragments: 1x v16h (2x b128) from 16-channel-grouped features.
// D: one 16B store per tile. Writes y+bias in the same grouped f16 layout.
// ---------------------------------------------------------------------------
template <int TM, int KP>
__global__ __launch_bounds__(256) void gemm_wmma_kernel(const _Float16* __restrict__ F,
                                                        const _Float16* __restrict__ W16,
                                                        const float* __restrict__ bias,
                                                        _Float16* __restrict__ Y)
{
    const int lane = threadIdx.x & 31;
    const int wave = threadIdx.x >> 5;
    const int tile_n = blockIdx.x * 8 + wave;       // grid sized exactly: tiles_n/8 blocks
    const int half = lane >> 4;
    const int mrow = lane & 15;
    const int ncol = tile_n * 16 + (lane & 15);

    v8f acc[TM] = {};
#pragma unroll
    for (int ks = 0; ks < KP; ks += 32) {
        // B fragment: K = ks + half*16 + (0..15), column ncol -> contiguous 32B
        const int kg = (ks >> 4) + half;
        const v16h bm = *(const v16h*)(F + ((size_t)kg * NTOT + ncol) * 16);
#pragma unroll
        for (int tm = 0; tm < TM; ++tm) {
            const _Float16* wrow = W16 + (size_t)(tm * 16 + mrow) * KP + ks + half * 16;
            const v8h alo = *(const v8h*)(wrow);        // K = ks+half*16   .. +7
            const v8h ahi = *(const v8h*)(wrow + 8);    // K = ks+half*16+8 .. +7
            v16h a;
#pragma unroll
            for (int i = 0; i < 8; ++i) { a[i] = alo[i]; a[i + 8] = ahi[i]; }
            acc[tm] = __builtin_amdgcn_wmma_f32_16x16x32_f16(
                false, a, false, bm, (short)0, acc[tm], false, false);
        }
    }
    // D layout: VGPR r -> channel tm*16 + half*8 + r, column ncol.
    // 8 consecutive channels inside group tm -> one 16B store.
#pragma unroll
    for (int tm = 0; tm < TM; ++tm) {
        v8h d;
#pragma unroll
        for (int r = 0; r < 8; ++r)
            d[r] = (_Float16)(acc[tm][r] + bias[tm * 16 + half * 8 + r]);
        *(v8h*)(Y + ((size_t)tm * NTOT + ncol) * 16 + half * 8) = d;
    }
}

// ---------------------------------------------------------------------------
// Kernel 5: per-channel mean / inv-std, one block per 16-channel group with
// fully coalesced v16h reads (each byte read exactly once).
// ---------------------------------------------------------------------------
__global__ __launch_bounds__(256) void stats_kernel(const _Float16* __restrict__ Y,
                                                    float* __restrict__ stats)
{
    const int kg = blockIdx.x;
    const int tid = threadIdx.x;
    float s0[16] = {}, s1[16] = {};
    for (size_t n = tid; n < NTOT; n += 256) {
        const v16h v = *(const v16h*)(Y + ((size_t)kg * NTOT + n) * 16);
#pragma unroll
        for (int t = 0; t < 16; ++t) {
            const float f = (float)v[t];
            s0[t] += f; s1[t] += f * f;
        }
    }
    __shared__ float r0[256], r1[256];
    for (int t = 0; t < 16; ++t) {
        r0[tid] = s0[t]; r1[tid] = s1[t];
        __syncthreads();
        for (int off = 128; off > 0; off >>= 1) {
            if (tid < off) { r0[tid] += r0[tid + off]; r1[tid] += r1[tid + off]; }
            __syncthreads();
        }
        if (tid == 0) {
            const float mean = r0[0] / (float)NTOT;
            const float var  = r1[0] / (float)NTOT - mean * mean;
            stats[2 * (kg * 16 + t)]     = mean;
            stats[2 * (kg * 16 + t) + 1] = rsqrtf(var + EPSV);
        }
        __syncthreads();
    }
}

// ---------------------------------------------------------------------------
// Kernel 6: BN + ReLU, one 16-half chunk (32B load + 32B store) per thread.
// Input and output share the grouped layout, so indices are identical.
// ---------------------------------------------------------------------------
__global__ __launch_bounds__(256) void bnrelu_kernel(const _Float16* __restrict__ Y,
                                                     const float* __restrict__ stats,
                                                     const float* __restrict__ g,
                                                     const float* __restrict__ be,
                                                     _Float16* __restrict__ Fout,
                                                     int C)
{
    const size_t chunk = (size_t)blockIdx.x * 256 + threadIdx.x;  // kg*NTOT + n
    if (chunk >= (size_t)(C / 16) * NTOT) return;
    const int kg = (int)(chunk / NTOT);
    const v16h v = *(const v16h*)(Y + chunk * 16);
    v16h o;
#pragma unroll
    for (int t = 0; t < 16; ++t) {
        const int c = kg * 16 + t;
        float f = (float)v[t];
        f = (f - stats[2 * c]) * stats[2 * c + 1] * g[c] + be[c];
        o[t] = (_Float16)fmaxf(f, 0.0f);
    }
    *(v16h*)(Fout + chunk * 16) = o;
}

// ---------------------------------------------------------------------------
// Kernel 7: fused BN + ReLU + max over NSAMPLE for the last layer.
// One thread per (b, o, s); output layout (B, C3, NPOINT).
// ---------------------------------------------------------------------------
__global__ __launch_bounds__(256) void final_kernel(const _Float16* __restrict__ Y,
                                                    const float* __restrict__ stats,
                                                    const float* __restrict__ g,
                                                    const float* __restrict__ be,
                                                    float* __restrict__ out)
{
    const size_t i = (size_t)blockIdx.x * 256 + threadIdx.x;  // b*C3*NPOINT + o*NPOINT + s
    if (i >= (size_t)NB * C3 * NPOINT) return;
    const int s = (int)(i % NPOINT);
    const int o = (int)((i / NPOINT) % C3);
    const int b = (int)(i / ((size_t)NPOINT * C3));
    const float mean = stats[2 * o], inv = stats[2 * o + 1];
    const float gg = g[o], bb = be[o];
    const _Float16* row = Y + ((size_t)(o >> 4) * NTOT) * 16 + (o & 15);
    const size_t nbase = ((size_t)b * NPOINT + s) * NSAMPLE;
    float m = 0.0f;   // relu output is >= 0
#pragma unroll
    for (int k = 0; k < NSAMPLE; ++k) {
        float v = (float)row[(nbase + k) * 16];
        v = fmaxf((v - mean) * inv * gg + bb, 0.0f);
        m = fmaxf(m, v);
    }
    out[i] = m;
}

// ---------------------------------------------------------------------------
extern "C" void kernel_launch(void* const* d_in, const int* in_sizes, int n_in,
                              void* d_out, int out_size, void* d_ws, size_t ws_size,
                              hipStream_t stream) {
    (void)in_sizes; (void)n_in; (void)out_size; (void)ws_size;
    const float* xyz = (const float*)d_in[0];
    const float* pts = (const float*)d_in[1];
    const float* w1  = (const float*)d_in[2];
    const float* b1  = (const float*)d_in[3];
    const float* g1  = (const float*)d_in[4];
    const float* be1 = (const float*)d_in[5];
    const float* w2  = (const float*)d_in[6];
    const float* b2  = (const float*)d_in[7];
    const float* g2  = (const float*)d_in[8];
    const float* be2 = (const float*)d_in[9];
    const float* w3  = (const float*)d_in[10];
    const float* b3  = (const float*)d_in[11];
    const float* g3  = (const float*)d_in[12];
    const float* be3 = (const float*)d_in[13];

    float* out      = (float*)d_out;
    float* out_xyz  = out;                                   // (B,3,NPOINT)
    float* out_feat = out + (size_t)NB * 3 * NPOINT;         // (B,C3,NPOINT)

    // workspace carve-up (256B aligned blocks)
    char* ws = (char*)d_ws;
    auto carve = [&](size_t bytes) { char* p = ws; ws += (bytes + 255) & ~(size_t)255; return p; };
    float*    new_xyz = (float*)   carve((size_t)NB * NPOINT * 3 * sizeof(float));
    float*    stats   = (float*)   carve(2 * C3 * sizeof(float));
    _Float16* w16a    = (_Float16*)carve((size_t)C1 * KPAD1 * sizeof(_Float16));
    _Float16* w16b    = (_Float16*)carve((size_t)C2 * C1 * sizeof(_Float16));
    _Float16* w16c    = (_Float16*)carve((size_t)C3 * C2 * sizeof(_Float16));
    _Float16* bufF    = (_Float16*)carve((size_t)KPAD1 * NTOT * sizeof(_Float16));
    _Float16* bufY    = (_Float16*)carve((size_t)C3 * NTOT * sizeof(_Float16));

    const int tiles_n = (int)(NTOT / 16);                    // 32768
    const int gemm_blocks = tiles_n / 8;                     // 8 waves per block

    // weight prep (tiny)
    wprep_kernel<<<(C1 * KPAD1 + 255) / 256, 256, 0, stream>>>(w1, w16a, C1, CIN1, KPAD1);
    wprep_kernel<<<(C2 * C1 + 255) / 256, 256, 0, stream>>>(w2, w16b, C2, C1, C1);
    wprep_kernel<<<(C3 * C2 + 255) / 256, 256, 0, stream>>>(w3, w16c, C3, C2, C2);

    // 1) FPS
    fps_kernel<<<NB, 256, 0, stream>>>(xyz, new_xyz, out_xyz);
    // 2) ball query + grouping into padded, 16-channel-grouped f16 features
    group_kernel<<<NB * NPOINT, 32, 0, stream>>>(xyz, pts, new_xyz, bufF);

    // 3) layer 1: (67->64), K padded to 96
    gemm_wmma_kernel<C1 / 16, KPAD1><<<gemm_blocks, 256, 0, stream>>>(bufF, w16a, b1, bufY);
    stats_kernel<<<C1 / 16, 256, 0, stream>>>(bufY, stats);
    bnrelu_kernel<<<(unsigned)(((size_t)(C1 / 16) * NTOT + 255) / 256), 256, 0, stream>>>(
        bufY, stats, g1, be1, bufF, C1);

    // 4) layer 2: (64->64)
    gemm_wmma_kernel<C2 / 16, C1><<<gemm_blocks, 256, 0, stream>>>(bufF, w16b, b2, bufY);
    stats_kernel<<<C2 / 16, 256, 0, stream>>>(bufY, stats);
    bnrelu_kernel<<<(unsigned)(((size_t)(C2 / 16) * NTOT + 255) / 256), 256, 0, stream>>>(
        bufY, stats, g2, be2, bufF, C2);

    // 5) layer 3: (64->128) + fused BN/ReLU/max-pool
    gemm_wmma_kernel<C3 / 16, C2><<<gemm_blocks, 256, 0, stream>>>(bufF, w16c, b3, bufY);
    stats_kernel<<<C3 / 16, 256, 0, stream>>>(bufY, stats);
    {
        const size_t tot = (size_t)NB * C3 * NPOINT;
        final_kernel<<<(unsigned)((tot + 255) / 256), 256, 0, stream>>>(bufY, stats, g3, be3, out_feat);
    }
}
